// GAT_37787122270994
// MI455X (gfx1250) — matrix operand order repaired and checked by reading, not compiled, not run on hardware
//
#include <hip/hip_runtime.h>

typedef _Float16 f16_t;
typedef __attribute__((ext_vector_type(16))) _Float16 v16h;
typedef __attribute__((ext_vector_type(8)))  float    v8f;

constexpr int N_NODE = 50000;
constexpr int N_EDGE = 500000;
constexpr int DIN    = 128;
constexpr int DHO    = 256;        // H * OUT
constexpr int NT     = DHO / 16;   // 16 column tiles

// ---------------------------------------------------------------------------
// Weight packing into WMMA B-fragment order.
// Packed halfs: index p = ((kt*NT + nt)*32 + lane)*16 + i
// value = W[kt*32 + (lane>>4)*8 + (i&7) + (i>>3)*16][nt*16 + (lane&15)]
// -> each lane's 16 halfs for a (kt,nt) tile are contiguous (32 bytes).
// ---------------------------------------------------------------------------
__global__ void pack_weight(const float* __restrict__ W, f16_t* __restrict__ out, int K) {
  int p = blockIdx.x * blockDim.x + threadIdx.x;
  int total = K * DHO;
  if (p >= total) return;
  int i    = p & 15;
  int lane = (p >> 4) & 31;
  int nt   = (p >> 9) & 15;
  int kt   = p >> 13;
  int k = kt * 32 + ((lane >> 4) * 8) + (i & 7) + ((i >> 3) * 16);
  int n = nt * 16 + (lane & 15);
  out[p] = (f16_t)W[k * DHO + n];
}

__global__ void fill_f32(float* __restrict__ p, long n, float v) {
  long i = (long)blockIdx.x * blockDim.x + threadIdx.x;
  if (i < n) p[i] = v;
}

// ---------------------------------------------------------------------------
// LDS staging of the A panel (16 rows x K), converted to f16 and stored in
// WMMA A-fragment order: sm[(kt*32 + lane)*16 + i]. One pass per block; all
// 4 waves then read their A operands from LDS (32B contiguous per lane).
// lda == K for every A matrix in this pipeline.
// ---------------------------------------------------------------------------
template <int K>
__device__ __forceinline__ void stage_a_f32(const float* __restrict__ A, int row0,
                                            f16_t* __restrict__ sm) {
  constexpr int NPAIR = (K / 32) * 32;
  for (int pair = threadIdx.x; pair < NPAIR; pair += 128) {
    int kt   = pair >> 5;
    int lane = pair & 31;
    int m    = row0 + (lane & 15);
    int base = (lane >> 4) * 8;
    const float* p = A + m * K + kt * 32 + base;
    const float4* q0 = (const float4*)p;          // 32B aligned
    const float4* q1 = (const float4*)(p + 16);
    float4 x0 = q0[0], x1 = q0[1], x2 = q1[0], x3 = q1[1];
    v16h a;
    a[0]=(f16_t)x0.x; a[1]=(f16_t)x0.y; a[2]=(f16_t)x0.z; a[3]=(f16_t)x0.w;
    a[4]=(f16_t)x1.x; a[5]=(f16_t)x1.y; a[6]=(f16_t)x1.z; a[7]=(f16_t)x1.w;
    a[8]=(f16_t)x2.x; a[9]=(f16_t)x2.y; a[10]=(f16_t)x2.z; a[11]=(f16_t)x2.w;
    a[12]=(f16_t)x3.x; a[13]=(f16_t)x3.y; a[14]=(f16_t)x3.z; a[15]=(f16_t)x3.w;
    *(v16h*)(sm + pair * 16) = a;                 // 32B LDS store
  }
  __syncthreads();
}

template <int K>
__device__ __forceinline__ void stage_a_f16(const f16_t* __restrict__ A, int row0,
                                            f16_t* __restrict__ sm) {
  constexpr int NPAIR = (K / 32) * 32;
  for (int pair = threadIdx.x; pair < NPAIR; pair += 128) {
    int kt   = pair >> 5;
    int lane = pair & 31;
    int m    = row0 + (lane & 15);
    int base = (lane >> 4) * 8;
    const f16_t* p = A + m * K + kt * 32 + base;
    v16h a;
#pragma unroll
    for (int i = 0; i < 8; ++i) a[i] = p[i];
#pragma unroll
    for (int i = 0; i < 8; ++i) a[8 + i] = p[16 + i];
    *(v16h*)(sm + pair * 16) = a;
  }
  __syncthreads();
}

__device__ __forceinline__ v16h lds_a(const f16_t* __restrict__ sm, int kt, int lane) {
  return *(const v16h*)(sm + ((kt * 32 + lane) << 4));   // 32B LDS load
}

__device__ __forceinline__ v16h load_b(const f16_t* __restrict__ Bp, int kt, int nt, int lane) {
  return *(const v16h*)(Bp + (((kt * NT + nt) * 32 + lane) << 4));
}

__device__ __forceinline__ void store_tile(float* __restrict__ C, int ldc,
                                           int row0, int col0, int lane, v8f acc) {
  int n  = col0 + (lane & 15);
  int m0 = row0 + ((lane >> 4) << 3);
#pragma unroll
  for (int r = 0; r < 8; ++r) C[(m0 + r) * ldc + n] = acc[r];
}

// ---------------------------------------------------------------------------
// Node projections: f_ni = X@Wni, f_nj = X@Wnj, h_src = X@Wsrc + b_src
// One LDS-staged A-tile feeds three WMMA accumulators x 4 waves.
// ---------------------------------------------------------------------------
__global__ void node_proj(const float* __restrict__ X,
                          const f16_t* __restrict__ Bni, const f16_t* __restrict__ Bnj,
                          const f16_t* __restrict__ Bsrc, const float* __restrict__ bias_src,
                          float* __restrict__ f_ni, float* __restrict__ f_nj,
                          float* __restrict__ h_src) {
  __shared__ __align__(32) f16_t sa[16 * DIN];
  int lane = threadIdx.x & 31;
  int wave = threadIdx.x >> 5;
  int nt = blockIdx.y * 4 + wave;
  int mt = blockIdx.x;
  stage_a_f32<DIN>(X, mt * 16, sa);
  v8f c0 = {}, c1 = {}, c2 = {};
#pragma unroll
  for (int kt = 0; kt < DIN / 32; ++kt) {
    v16h a  = lds_a(sa, kt, lane);
    v16h b0 = load_b(Bni,  kt, nt, lane);
    v16h b1 = load_b(Bnj,  kt, nt, lane);
    v16h b2 = load_b(Bsrc, kt, nt, lane);
    c0 = __builtin_amdgcn_wmma_f32_16x16x32_f16(false, a, false, b0, (short)0, c0, false, false);
    c1 = __builtin_amdgcn_wmma_f32_16x16x32_f16(false, a, false, b1, (short)0, c1, false, false);
    c2 = __builtin_amdgcn_wmma_f32_16x16x32_f16(false, a, false, b2, (short)0, c2, false, false);
  }
  store_tile(f_ni, DHO, mt * 16, nt * 16, lane, c0);
  store_tile(f_nj, DHO, mt * 16, nt * 16, lane, c1);
  float bs = bias_src[nt * 16 + (lane & 15)];
#pragma unroll
  for (int r = 0; r < 8; ++r) c2[r] += bs;
  store_tile(h_src, DHO, mt * 16, nt * 16, lane, c2);
}

// ---------------------------------------------------------------------------
// Edge features: f_out = leaky_relu(EF@Wfij + f_ni[src] + f_nj[dst]), stored f16
// ---------------------------------------------------------------------------
__global__ void edge_feat(const float* __restrict__ EF, const f16_t* __restrict__ Bfij,
                          const float* __restrict__ f_ni, const float* __restrict__ f_nj,
                          const int* __restrict__ src, const int* __restrict__ dst,
                          f16_t* __restrict__ f_out) {
  __shared__ __align__(32) f16_t sa[16 * DIN];
  int lane = threadIdx.x & 31;
  int wave = threadIdx.x >> 5;
  int nt = blockIdx.y * 4 + wave;
  int mt = blockIdx.x;
  stage_a_f32<DIN>(EF, mt * 16, sa);
  v8f c = {};
#pragma unroll
  for (int kt = 0; kt < DIN / 32; ++kt) {
    v16h a = lds_a(sa, kt, lane);
    v16h b = load_b(Bfij, kt, nt, lane);
    c = __builtin_amdgcn_wmma_f32_16x16x32_f16(false, a, false, b, (short)0, c, false, false);
  }
  int n  = nt * 16 + (lane & 15);
  int m0 = mt * 16 + ((lane >> 4) << 3);
#pragma unroll
  for (int r = 0; r < 8; ++r) {
    int row = m0 + r;
    int s = src[row], d = dst[row];
    float v = c[r] + f_ni[s * DHO + n] + f_nj[d * DHO + n];
    v = v > 0.f ? v : 0.01f * v;
    f_out[row * DHO + n] = (f16_t)v;
  }
}

// ---------------------------------------------------------------------------
// Attention logits e[E][4] + segment max via float-as-int atomic max
// ---------------------------------------------------------------------------
__global__ void attn_logits(const f16_t* __restrict__ f_out, const float* __restrict__ attn,
                            const int* __restrict__ dst, float* __restrict__ e_buf,
                            float* __restrict__ seg_max) {
  int gid = blockIdx.x * blockDim.x + threadIdx.x;
  if (gid >= N_EDGE * 4) return;
  int edge = gid >> 2;
  int h    = gid & 3;
  const v16h* pv = (const v16h*)(f_out + edge * DHO + h * 64);  // 128B aligned
  const float* ar = attn + h * 64;
  float s = 0.f;
#pragma unroll
  for (int ch = 0; ch < 4; ++ch) {
    v16h x = pv[ch];
#pragma unroll
    for (int i = 0; i < 16; ++i) s += (float)x[i] * ar[ch * 16 + i];
  }
  e_buf[gid] = s;
  float* addr = seg_max + dst[edge] * 4 + h;
  if (s >= 0.f) atomicMax((int*)addr, __float_as_int(s));
  else          atomicMin((unsigned int*)addr, (unsigned int)__float_as_int(s));
}

__global__ void softmax_ex(float* __restrict__ e_buf, const float* __restrict__ seg_max,
                           const int* __restrict__ dst, float* __restrict__ den) {
  int gid = blockIdx.x * blockDim.x + threadIdx.x;
  if (gid >= N_EDGE * 4) return;
  int edge = gid >> 2;
  int h    = gid & 3;
  int d = dst[edge];
  float x = expf(e_buf[gid] - seg_max[d * 4 + h]);
  e_buf[gid] = x;                // overwrite logits with exp values
  atomicAdd(den + d * 4 + h, x);
}

// one block per edge, 256 threads = 256 channels
__global__ void scatter_msg(const float* __restrict__ h_src, const float* __restrict__ ex,
                            const float* __restrict__ den, const int* __restrict__ src,
                            const int* __restrict__ dst, float* __restrict__ h_out) {
  int edge = blockIdx.x;
  int t = threadIdx.x;
  int h = t >> 6;
  int s = src[edge], d = dst[edge];
  float a = ex[edge * 4 + h] / den[d * 4 + h];
  atomicAdd(h_out + d * DHO + t, h_src[s * DHO + t] * a);
}

// ---------------------------------------------------------------------------
// Dense (K=256) GEMM + bias + ReLU, f32 or f16 A operand, LDS-staged A
// ---------------------------------------------------------------------------
__global__ void dense_relu_f32(const float* __restrict__ A, const f16_t* __restrict__ Bw,
                               const float* __restrict__ bias, float* __restrict__ out) {
  __shared__ __align__(32) f16_t sa[16 * DHO];
  int lane = threadIdx.x & 31;
  int wave = threadIdx.x >> 5;
  int nt = blockIdx.y * 4 + wave;
  int mt = blockIdx.x;
  stage_a_f32<DHO>(A, mt * 16, sa);
  v8f c = {};
#pragma unroll
  for (int kt = 0; kt < DHO / 32; ++kt) {
    v16h a = lds_a(sa, kt, lane);
    v16h b = load_b(Bw, kt, nt, lane);
    c = __builtin_amdgcn_wmma_f32_16x16x32_f16(false, a, false, b, (short)0, c, false, false);
  }
  int n  = nt * 16 + (lane & 15);
  int m0 = mt * 16 + ((lane >> 4) << 3);
  float bb = bias[n];
#pragma unroll
  for (int r = 0; r < 8; ++r) {
    float v = c[r] + bb;
    out[(m0 + r) * DHO + n] = v > 0.f ? v : 0.f;
  }
}

__global__ void dense_relu_f16(const f16_t* __restrict__ A, const f16_t* __restrict__ Bw,
                               const float* __restrict__ bias, float* __restrict__ out) {
  __shared__ __align__(32) f16_t sa[16 * DHO];
  int lane = threadIdx.x & 31;
  int wave = threadIdx.x >> 5;
  int nt = blockIdx.y * 4 + wave;
  int mt = blockIdx.x;
  stage_a_f16<DHO>(A, mt * 16, sa);
  v8f c = {};
#pragma unroll
  for (int kt = 0; kt < DHO / 32; ++kt) {
    v16h a = lds_a(sa, kt, lane);
    v16h b = load_b(Bw, kt, nt, lane);
    c = __builtin_amdgcn_wmma_f32_16x16x32_f16(false, a, false, b, (short)0, c, false, false);
  }
  int n  = nt * 16 + (lane & 15);
  int m0 = mt * 16 + ((lane >> 4) << 3);
  float bb = bias[n];
#pragma unroll
  for (int r = 0; r < 8; ++r) {
    float v = c[r] + bb;
    out[(m0 + r) * DHO + n] = v > 0.f ? v : 0.f;
  }
}

// ---------------------------------------------------------------------------
// In-place LayerNorm over 256-wide rows (one block per row)
// ---------------------------------------------------------------------------
__global__ void layer_norm(float* __restrict__ data, const float* __restrict__ g,
                           const float* __restrict__ b) {
  long row = blockIdx.x;
  float* p = data + row * (long)DHO;
  int t = threadIdx.x;
  float v = p[t];
  __shared__ float red[DHO];
  red[t] = v;
  __syncthreads();
  for (int s = DHO / 2; s > 0; s >>= 1) {
    if (t < s) red[t] += red[t + s];
    __syncthreads();
  }
  float mu = red[0] * (1.f / DHO);
  __syncthreads();
  float dv = v - mu;
  red[t] = dv * dv;
  __syncthreads();
  for (int s = DHO / 2; s > 0; s >>= 1) {
    if (t < s) red[t] += red[t + s];
    __syncthreads();
  }
  float var = red[0] * (1.f / DHO);
  p[t] = dv * rsqrtf(var + 1e-5f) * g[t] + b[t];
}

// ---------------------------------------------------------------------------
extern "C" void kernel_launch(void* const* d_in, const int* in_sizes, int n_in,
                              void* d_out, int out_size, void* d_ws, size_t ws_size,
                              hipStream_t stream) {
  (void)in_sizes; (void)n_in; (void)out_size; (void)ws_size;

  const float* node_feature = (const float*)d_in[0];
  const float* edge_feature = (const float*)d_in[1];
  const float* W_ni     = (const float*)d_in[2];
  const float* W_nj     = (const float*)d_in[3];
  const float* W_fij    = (const float*)d_in[4];
  const float* attn     = (const float*)d_in[5];
  const float* W_src    = (const float*)d_in[6];
  const float* b_src    = (const float*)d_in[7];
  const float* W_edense = (const float*)d_in[8];
  const float* b_edense = (const float*)d_in[9];
  const float* W_ndense = (const float*)d_in[10];
  const float* b_ndense = (const float*)d_in[11];
  const float* ln_g     = (const float*)d_in[12];
  const float* ln_b     = (const float*)d_in[13];
  const int*   src      = (const int*)d_in[14];
  const int*   dst      = (const int*)d_in[15];

  // ---- workspace carve-up (256B aligned) ----
  char* w = (char*)d_ws;
  auto carve = [&](size_t bytes) {
    char* p = w;
    w += (bytes + 255) & ~(size_t)255;
    return p;
  };
  f16_t* wf_ni   = (f16_t*)carve((size_t)DIN * DHO * 2);
  f16_t* wf_nj   = (f16_t*)carve((size_t)DIN * DHO * 2);
  f16_t* wf_fij  = (f16_t*)carve((size_t)DIN * DHO * 2);
  f16_t* wf_src  = (f16_t*)carve((size_t)DIN * DHO * 2);
  f16_t* wf_ed   = (f16_t*)carve((size_t)DHO * DHO * 2);
  f16_t* wf_nd   = (f16_t*)carve((size_t)DHO * DHO * 2);
  float* f_ni    = (float*)carve((size_t)N_NODE * DHO * 4);
  float* f_nj    = (float*)carve((size_t)N_NODE * DHO * 4);
  float* h_src   = (float*)carve((size_t)N_NODE * DHO * 4);
  float* h_out   = (float*)carve((size_t)N_NODE * DHO * 4);
  float* e_buf   = (float*)carve((size_t)N_EDGE * 4 * 4);
  float* seg_max = (float*)carve((size_t)N_NODE * 4 * 4);
  float* den     = (float*)carve((size_t)N_NODE * 4 * 4);
  f16_t* f_out16 = (f16_t*)carve((size_t)N_EDGE * DHO * 2);

  float* out_node = (float*)d_out;
  float* out_edge = out_node + (size_t)N_NODE * DHO;

  // ---- 1. pack weights to WMMA-B layout ----
  pack_weight<<<(DIN * DHO + 255) / 256, 256, 0, stream>>>(W_ni,     wf_ni,  DIN);
  pack_weight<<<(DIN * DHO + 255) / 256, 256, 0, stream>>>(W_nj,     wf_nj,  DIN);
  pack_weight<<<(DIN * DHO + 255) / 256, 256, 0, stream>>>(W_fij,    wf_fij, DIN);
  pack_weight<<<(DIN * DHO + 255) / 256, 256, 0, stream>>>(W_src,    wf_src, DIN);
  pack_weight<<<(DHO * DHO + 255) / 256, 256, 0, stream>>>(W_edense, wf_ed,  DHO);
  pack_weight<<<(DHO * DHO + 255) / 256, 256, 0, stream>>>(W_ndense, wf_nd,  DHO);

  // ---- 2. init accumulation buffers ----
  fill_f32<<<(N_NODE * 4 + 255) / 256, 256, 0, stream>>>(seg_max, (long)N_NODE * 4,
                                                         -__builtin_inff());
  fill_f32<<<(N_NODE * 4 + 255) / 256, 256, 0, stream>>>(den, (long)N_NODE * 4, 0.f);
  fill_f32<<<(N_NODE * DHO + 255) / 256, 256, 0, stream>>>(h_out, (long)N_NODE * DHO, 0.f);

  // ---- 3. node projections (WMMA, LDS-staged A) ----
  node_proj<<<dim3(N_NODE / 16, 4), 128, 0, stream>>>(node_feature, wf_ni, wf_nj, wf_src,
                                                      b_src, f_ni, f_nj, h_src);

  // ---- 4. edge features (WMMA + gather + leaky relu) ----
  edge_feat<<<dim3(N_EDGE / 16, 4), 128, 0, stream>>>(edge_feature, wf_fij, f_ni, f_nj,
                                                      src, dst, f_out16);

  // ---- 5. attention: logits, segment max, exp, segment sum ----
  attn_logits<<<(N_EDGE * 4 + 255) / 256, 256, 0, stream>>>(f_out16, attn, dst, e_buf, seg_max);
  softmax_ex<<<(N_EDGE * 4 + 255) / 256, 256, 0, stream>>>(e_buf, seg_max, dst, den);

  // ---- 6. attention-weighted scatter aggregation ----
  scatter_msg<<<N_EDGE, 256, 0, stream>>>(h_src, e_buf, den, src, dst, h_out);

  // ---- 7. trailing dense layers (WMMA) straight into d_out ----
  dense_relu_f32<<<dim3(N_NODE / 16, 4), 128, 0, stream>>>(h_out, wf_nd, b_ndense, out_node);
  dense_relu_f16<<<dim3(N_EDGE / 16, 4), 128, 0, stream>>>(f_out16, wf_ed, b_edense, out_edge);

  // ---- 8. in-place LayerNorm over both outputs ----
  layer_norm<<<N_NODE + N_EDGE, 256, 0, stream>>>(out_node, ln_g, ln_b);
}
